// GAT_GCN_72868415144433
// MI455X (gfx1250) — matrix operand order repaired and checked by reading, not compiled, not run on hardware
//
#include <hip/hip_runtime.h>
#include <hip/hip_bf16.h>

// ---------------- problem constants ----------------
#define NN    30000      // nodes
#define NNP   30016      // nodes padded to mult of 32 (A-row alloc for GEMM)
#define NE    240000     // original edges
#define ET    270000     // edges + self loops
#define NG    600        // graphs
#define FI    78         // in features
#define NH    10         // heads
#define FH    780        // hidden = FI*NH
#define SLOPE 0.2f

// padded dims
#define K1P   96         // FI  -> pad to mult of 32
#define N1P   800        // FH + 2*NH alphas (exactly 50 tiles)
#define N1A   832        // N1P rounded to 64 (Bt row alloc)
#define K2P   800        // FH -> 800
#define N2A   832        // 780 -> 832 rows alloc
#define XWLD  784        // xw row stride (780 -> 784)
#define MFC   608        // 600 -> 608 rows (mult of 32)
#define KF1   1568       // 1560 -> 1568
#define NF1A  1536       // 1500 -> 1536 rows alloc
#define KF2   1504       // 1500 -> 1504
#define NF2A  128        // 128 (2 tiles of 64)

typedef __attribute__((ext_vector_type(16))) __bf16 v16bf;
typedef __attribute__((ext_vector_type(8)))  float  v8f;

// ================= generic bf16 WMMA GEMM =================
// C[M,N] (f32, opt bias/relu) = A[M,K]bf16 row-major  x  Bt[Nalloc,K]bf16 (B transposed)
// wave tile 32x64 (2 A-frags x 4 B-tiles = 8 WMMAs, B reused across M-halves)
// block = 8 waves stacked in M => 256x64. M_alloc must be a multiple of 32.
__global__ __launch_bounds__(256)
void gemm_bf16_wmma(const __bf16* __restrict__ A, const __bf16* __restrict__ Bt,
                    float* __restrict__ C, const float* __restrict__ bias,
                    int M_alloc, int M_valid, int N_valid,
                    int K, int lda, int ldb, int ldc, int relu)
{
    const int wave = threadIdx.x >> 5;
    const int lane = threadIdx.x & 31;
    const int r    = lane & 15;
    const int hi   = lane >> 4;          // upper half-wave

    const int m0 = blockIdx.y * 256 + wave * 32;
    const int n0 = blockIdx.x * 64;
    if (m0 >= M_alloc) return;

    const __bf16* Arow0 = A + (size_t)(m0 + r) * lda;
    const __bf16* Arow1 = Arow0 + (size_t)16 * lda;
    const __bf16* Bp0   = Bt + (size_t)(n0 + r) * ldb;
    const __bf16* Bp1   = Bp0 + (size_t)16 * ldb;
    const __bf16* Bp2   = Bp0 + (size_t)32 * ldb;
    const __bf16* Bp3   = Bp0 + (size_t)48 * ldb;

    v8f acc0 = {}, acc1 = {}, acc2 = {}, acc3 = {};   // rows m0..m0+15
    v8f acc4 = {}, acc5 = {}, acc6 = {}, acc7 = {};   // rows m0+16..m0+31

    for (int k0 = 0; k0 < K; k0 += 32) {
        union { v16bf v; uint4 q[2]; } a0, a1, b0, b1, b2, b3;
        // A 16x32 bf16 fragment: lanes0-15 K 0..7 / 16..23 ; lanes16-31 K 8..15 / 24..31
        const int ka = k0 + hi * 8;
        a0.q[0] = *(const uint4*)(Arow0 + ka);
        a0.q[1] = *(const uint4*)(Arow0 + ka + 16);
        a1.q[0] = *(const uint4*)(Arow1 + ka);
        a1.q[1] = *(const uint4*)(Arow1 + ka + 16);
        // B 32x16 fragment (from transposed weights): lanes0-15 K 0..7 / 16..23 ; lanes16-31 K 8..15 / 24..31
        const int kb = k0 + hi * 8;
        b0.q[0] = *(const uint4*)(Bp0 + kb); b0.q[1] = *(const uint4*)(Bp0 + kb + 16);
        b1.q[0] = *(const uint4*)(Bp1 + kb); b1.q[1] = *(const uint4*)(Bp1 + kb + 16);
        b2.q[0] = *(const uint4*)(Bp2 + kb); b2.q[1] = *(const uint4*)(Bp2 + kb + 16);
        b3.q[0] = *(const uint4*)(Bp3 + kb); b3.q[1] = *(const uint4*)(Bp3 + kb + 16);

        acc0 = __builtin_amdgcn_wmma_f32_16x16x32_bf16(false, a0.v, false, b0.v, (short)0, acc0, false, false);
        acc4 = __builtin_amdgcn_wmma_f32_16x16x32_bf16(false, a1.v, false, b0.v, (short)0, acc4, false, false);
        acc1 = __builtin_amdgcn_wmma_f32_16x16x32_bf16(false, a0.v, false, b1.v, (short)0, acc1, false, false);
        acc5 = __builtin_amdgcn_wmma_f32_16x16x32_bf16(false, a1.v, false, b1.v, (short)0, acc5, false, false);
        acc2 = __builtin_amdgcn_wmma_f32_16x16x32_bf16(false, a0.v, false, b2.v, (short)0, acc2, false, false);
        acc6 = __builtin_amdgcn_wmma_f32_16x16x32_bf16(false, a1.v, false, b2.v, (short)0, acc6, false, false);
        acc3 = __builtin_amdgcn_wmma_f32_16x16x32_bf16(false, a0.v, false, b3.v, (short)0, acc3, false, false);
        acc7 = __builtin_amdgcn_wmma_f32_16x16x32_bf16(false, a1.v, false, b3.v, (short)0, acc7, false, false);
    }

    v8f accs[8] = {acc0, acc1, acc2, acc3, acc4, acc5, acc6, acc7};
#pragma unroll
    for (int half = 0; half < 2; ++half) {
#pragma unroll
        for (int t = 0; t < 4; ++t) {
            const int col = n0 + t * 16 + r;
            if (col >= N_valid) continue;
            const float bv = bias ? bias[col] : 0.0f;
#pragma unroll
            for (int v = 0; v < 8; ++v) {
                const int row = m0 + half * 16 + v + hi * 8;  // C/D: VGPR v -> M=v (+8 hi half)
                if (row < M_valid) {
                    float x = accs[half * 4 + t][v] + bv;
                    if (relu) x = fmaxf(x, 0.0f);
                    C[(size_t)row * ldc + col] = x;
                }
            }
        }
    }
}

// ================= helpers / elementwise =================
__global__ void fill_f32(float* p, float v, int n) {
    int i = blockIdx.x * blockDim.x + threadIdx.x;
    if (i < n) p[i] = v;
}

__global__ void conv_x_bf16(const float* __restrict__ x, __bf16* __restrict__ xb) {
    int i = blockIdx.x * blockDim.x + threadIdx.x;
    if (i >= NNP * K1P) return;
    int n = i / K1P, k = i - n * K1P;
    xb[i] = (n < NN && k < FI) ? (__bf16)x[(size_t)n * FI + k] : (__bf16)0.0f;
}

// gat weights transposed + attention vectors folded into rows 780..799
__global__ void conv_wgat(const float* __restrict__ W, const float* __restrict__ asrc,
                          const float* __restrict__ adst, __bf16* __restrict__ Bt) {
    int i = blockIdx.x * blockDim.x + threadIdx.x;
    if (i >= N1A * K1P) return;
    int n = i / K1P, k = i - n * K1P;
    float v = 0.0f;
    if (k < FI) {
        if (n < FH) {
            v = W[(size_t)k * FH + n];
        } else if (n < FH + NH) {
            int h = n - FH;
            float s = 0.0f;
            for (int f = 0; f < FI; ++f) s += W[(size_t)k * FH + h * FI + f] * asrc[h * FI + f];
            v = s;
        } else if (n < FH + 2 * NH) {
            int h = n - FH - NH;
            float s = 0.0f;
            for (int f = 0; f < FI; ++f) s += W[(size_t)k * FH + h * FI + f] * adst[h * FI + f];
            v = s;
        }
    }
    Bt[i] = (__bf16)v;
}

// generic transpose+pad f32 [K,N] -> bf16 [Nalloc,Kpad]
__global__ void wtrans(const float* __restrict__ W, __bf16* __restrict__ Bt,
                       int K, int N, int Kpad, int Nalloc) {
    int i = blockIdx.x * blockDim.x + threadIdx.x;
    if (i >= Nalloc * Kpad) return;
    int n = i / Kpad, k = i - n * Kpad;
    Bt[i] = (n < N && k < K) ? (__bf16)W[(size_t)k * N + n] : (__bf16)0.0f;
}

__device__ __forceinline__ void edge_sd(const int* __restrict__ ei, int e, int& s, int& d) {
    if (e < NE) { s = ei[e]; d = ei[NE + e]; }
    else        { s = d = e - NE; }          // self loops appended
}

__device__ __forceinline__ void atomicMaxF(float* addr, float v) {
    if (v >= 0.0f) atomicMax((int*)addr, __float_as_int(v));
    else           atomicMin((unsigned int*)addr, __float_as_uint(v));
}

__global__ void edge_max(const int* __restrict__ ei, const float* __restrict__ HA,
                         float* __restrict__ mx) {
    int i = blockIdx.x * blockDim.x + threadIdx.x;
    if (i >= ET * NH) return;
    int e = i / NH, h = i - e * NH;
    int s, d; edge_sd(ei, e, s, d);
    float v = HA[(size_t)s * N1P + FH + h] + HA[(size_t)d * N1P + FH + NH + h];
    v = (v > 0.0f) ? v : SLOPE * v;
    atomicMaxF(&mx[d * NH + h], v);
}

__global__ void edge_exp(const int* __restrict__ ei, const float* __restrict__ HA,
                         const float* __restrict__ mx, float* __restrict__ ae,
                         float* __restrict__ den) {
    int i = blockIdx.x * blockDim.x + threadIdx.x;
    if (i >= ET * NH) return;
    int e = i / NH, h = i - e * NH;
    int s, d; edge_sd(ei, e, s, d);
    float v = HA[(size_t)s * N1P + FH + h] + HA[(size_t)d * N1P + FH + NH + h];
    v = (v > 0.0f) ? v : SLOPE * v;
    float a = __expf(v - mx[d * NH + h]);
    ae[i] = a;
    atomicAdd(&den[d * NH + h], a);
}

// one block per edge: acc[d] += softmax_w * h[s]
__global__ __launch_bounds__(256)
void edge_scatter_gat(const int* __restrict__ ei, const float* __restrict__ HA,
                      const float* __restrict__ ae, const float* __restrict__ den,
                      float* __restrict__ acc) {
    __shared__ float w[NH];
    int e = blockIdx.x;
    int s, d; edge_sd(ei, e, s, d);
    if (threadIdx.x < NH)
        w[threadIdx.x] = ae[(size_t)e * NH + threadIdx.x] /
                         fmaxf(den[d * NH + threadIdx.x], 1e-38f);
    __syncthreads();
    const float* hs = HA + (size_t)s * N1P;
    float* od = acc + (size_t)d * FH;
    for (int c = threadIdx.x; c < FH; c += 256)
        atomicAdd(&od[c], w[c / FI] * hs[c]);
}

// relu(acc + bias) -> bf16 padded [NNP, K2P]
__global__ void conv_h1(const float* __restrict__ acc, const float* __restrict__ bias,
                        __bf16* __restrict__ h1b) {
    int i = blockIdx.x * blockDim.x + threadIdx.x;
    if (i >= NNP * K2P) return;
    int n = i / K2P, c = i - n * K2P;
    float v = 0.0f;
    if (n < NN && c < FH) v = fmaxf(acc[(size_t)n * FH + c] + bias[c], 0.0f);
    h1b[i] = (__bf16)v;
}

__global__ void edge_deg(const int* __restrict__ ei, float* __restrict__ deg) {
    int e = blockIdx.x * blockDim.x + threadIdx.x;
    if (e >= ET) return;
    int s, d; edge_sd(ei, e, s, d);
    atomicAdd(&deg[d], 1.0f);
}

__global__ void calc_dinv(const float* __restrict__ deg, float* __restrict__ dinv) {
    int i = blockIdx.x * blockDim.x + threadIdx.x;
    if (i < NN) dinv[i] = rsqrtf(fmaxf(deg[i], 1.0f));
}

__global__ __launch_bounds__(256)
void edge_scatter_gcn(const int* __restrict__ ei, const float* __restrict__ xw,
                      const float* __restrict__ dinv, float* __restrict__ acc) {
    int e = blockIdx.x;
    int s, d; edge_sd(ei, e, s, d);
    const float coef = dinv[s] * dinv[d];
    const float* xs = xw + (size_t)s * XWLD;
    float* od = acc + (size_t)d * FH;
    for (int c = threadIdx.x; c < FH; c += 256)
        atomicAdd(&od[c], coef * xs[c]);
}

// block per node: relu(acc + bias) pooled into graphs
__global__ __launch_bounds__(256)
void pool_nodes(const float* __restrict__ acc, const float* __restrict__ bias,
                const int* __restrict__ batch, float* __restrict__ pool,
                float* __restrict__ cnt) {
    int n = blockIdx.x;
    int g = batch[n];
    const float* an = acc + (size_t)n * FH;
    float* pg = pool + (size_t)g * FH;
    for (int c = threadIdx.x; c < FH; c += 256)
        atomicAdd(&pg[c], fmaxf(an[c] + bias[c], 0.0f));
    if (threadIdx.x == 0) atomicAdd(&cnt[g], 1.0f);
}

// build [MFC, KF1] bf16: concat(mean_pool, add_pool), zero-padded
__global__ void build_g(const float* __restrict__ pool, const float* __restrict__ cnt,
                        __bf16* __restrict__ gb) {
    int i = blockIdx.x * blockDim.x + threadIdx.x;
    if (i >= MFC * KF1) return;
    int gi = i / KF1, c = i - gi * KF1;
    float v = 0.0f;
    if (gi < NG) {
        if (c < FH)            v = pool[(size_t)gi * FH + c] / fmaxf(cnt[gi], 1.0f);
        else if (c < 2 * FH)   v = pool[(size_t)gi * FH + (c - FH)];
    }
    gb[i] = (__bf16)v;
}

// f32 [600,1500] -> bf16 [MFC, KF2] padded
__global__ void conv_fc1o(const float* __restrict__ src, __bf16* __restrict__ dst) {
    int i = blockIdx.x * blockDim.x + threadIdx.x;
    if (i >= MFC * KF2) return;
    int m = i / KF2, k = i - m * KF2;
    float v = (m < NG && k < 1500) ? src[(size_t)m * 1500 + k] : 0.0f;
    dst[i] = (__bf16)v;
}

// ================= launcher =================
static inline size_t al256(size_t x) { return (x + 255) & ~(size_t)255; }

extern "C" void kernel_launch(void* const* d_in, const int* in_sizes, int n_in,
                              void* d_out, int out_size, void* d_ws, size_t ws_size,
                              hipStream_t stream) {
    (void)in_sizes; (void)n_in; (void)out_size; (void)ws_size;
    const float* x      = (const float*)d_in[0];
    const int*   ei     = (const int*)  d_in[1];
    const int*   batch  = (const int*)  d_in[2];
    const float* gat_w  = (const float*)d_in[3];
    const float* a_src  = (const float*)d_in[4];
    const float* a_dst  = (const float*)d_in[5];
    const float* gat_b  = (const float*)d_in[6];
    const float* gcn_w  = (const float*)d_in[7];
    const float* gcn_b  = (const float*)d_in[8];
    const float* fc1_w  = (const float*)d_in[9];
    const float* fc1_b  = (const float*)d_in[10];
    const float* fc2_w  = (const float*)d_in[11];
    const float* fc2_b  = (const float*)d_in[12];
    float* out = (float*)d_out;

    // ---- carve workspace ----
    char* base = (char*)d_ws; size_t off = 0;
    auto carve = [&](size_t bytes) { char* r = base + off; off = al256(off + bytes); return r; };
    __bf16* XB    = (__bf16*)carve((size_t)NNP * K1P * 2);
    __bf16* WB1   = (__bf16*)carve((size_t)N1A * K1P * 2);
    float*  HA    = (float*) carve((size_t)NN * N1P * 4);   // later reused as XW [NN,XWLD]
    float*  GACC  = (float*) carve((size_t)NN * FH * 4);    // reused for GCN acc
    __bf16* H1B   = (__bf16*)carve((size_t)NNP * K2P * 2);
    __bf16* WB2   = (__bf16*)carve((size_t)N2A * K2P * 2);
    float*  MAXB  = (float*) carve((size_t)NN * NH * 4);
    float*  DEN   = (float*) carve((size_t)NN * NH * 4);
    float*  AE    = (float*) carve((size_t)ET * NH * 4);
    float*  DEG   = (float*) carve((size_t)NN * 4);
    float*  DINV  = (float*) carve((size_t)NN * 4);
    float*  POOL  = (float*) carve((size_t)NG * FH * 4);
    float*  CNT   = (float*) carve((size_t)NG * 4);
    __bf16* GB    = (__bf16*)carve((size_t)MFC * KF1 * 2);
    __bf16* WF1   = (__bf16*)carve((size_t)NF1A * KF1 * 2);
    float*  FC1O  = (float*) carve((size_t)NG * 1500 * 4);
    __bf16* G2B   = (__bf16*)carve((size_t)MFC * KF2 * 2);
    __bf16* WF2   = (__bf16*)carve((size_t)NF2A * KF2 * 2);
    float*  XW    = HA;  // reuse after GAT phase

    const int TB = 256;
    auto blocks = [](long n, int tb) { return (unsigned)((n + tb - 1) / tb); };

    // ---- GAT linear (+folded attention columns) ----
    conv_x_bf16<<<blocks((long)NNP * K1P, TB), TB, 0, stream>>>(x, XB);
    conv_wgat  <<<blocks((long)N1A * K1P, TB), TB, 0, stream>>>(gat_w, a_src, a_dst, WB1);
    {
        dim3 g((N1P + 63) / 64, (NNP + 255) / 256);
        gemm_bf16_wmma<<<g, 256, 0, stream>>>(XB, WB1, HA, nullptr,
                                              NNP, NN, N1P, K1P, K1P, K1P, N1P, 0);
    }

    // ---- edge attention softmax ----
    fill_f32<<<blocks((long)NN * NH, TB), TB, 0, stream>>>(MAXB, -3.402823466e38f, NN * NH);
    hipMemsetAsync(DEN, 0, (size_t)NN * NH * 4, stream);
    edge_max<<<blocks((long)ET * NH, TB), TB, 0, stream>>>(ei, HA, MAXB);
    edge_exp<<<blocks((long)ET * NH, TB), TB, 0, stream>>>(ei, HA, MAXB, AE, DEN);

    // ---- GAT aggregate ----
    hipMemsetAsync(GACC, 0, (size_t)NN * FH * 4, stream);
    edge_scatter_gat<<<ET, 256, 0, stream>>>(ei, HA, AE, DEN, GACC);
    conv_h1<<<blocks((long)NNP * K2P, TB), TB, 0, stream>>>(GACC, gat_b, H1B);

    // ---- GCN linear ----
    wtrans<<<blocks((long)N2A * K2P, TB), TB, 0, stream>>>(gcn_w, WB2, FH, FH, K2P, N2A);
    {
        dim3 g((FH + 63) / 64, (NNP + 255) / 256);
        gemm_bf16_wmma<<<g, 256, 0, stream>>>(H1B, WB2, XW, nullptr,
                                              NNP, NN, FH, K2P, K2P, K2P, XWLD, 0);
    }

    // ---- GCN normalize + aggregate ----
    hipMemsetAsync(DEG, 0, (size_t)NN * 4, stream);
    edge_deg<<<blocks(ET, TB), TB, 0, stream>>>(ei, DEG);
    calc_dinv<<<blocks(NN, TB), TB, 0, stream>>>(DEG, DINV);
    hipMemsetAsync(GACC, 0, (size_t)NN * FH * 4, stream);
    edge_scatter_gcn<<<ET, 256, 0, stream>>>(ei, XW, DINV, GACC);

    // ---- pooling ----
    hipMemsetAsync(POOL, 0, (size_t)NG * FH * 4, stream);
    hipMemsetAsync(CNT, 0, (size_t)NG * 4, stream);
    pool_nodes<<<NN, 256, 0, stream>>>(GACC, gcn_b, batch, POOL, CNT);
    build_g<<<blocks((long)MFC * KF1, TB), TB, 0, stream>>>(POOL, CNT, GB);

    // ---- fc1 (relu) ----
    wtrans<<<blocks((long)NF1A * KF1, TB), TB, 0, stream>>>(fc1_w, WF1, 1560, 1500, KF1, NF1A);
    {
        dim3 g((1500 + 63) / 64, (MFC + 255) / 256);
        gemm_bf16_wmma<<<g, 256, 0, stream>>>(GB, WF1, FC1O, fc1_b,
                                              MFC, NG, 1500, KF1, KF1, KF1, 1500, 1);
    }
    conv_fc1o<<<blocks((long)MFC * KF2, TB), TB, 0, stream>>>(FC1O, G2B);

    // ---- fc2 -> out ----
    wtrans<<<blocks((long)NF2A * KF2, TB), TB, 0, stream>>>(fc2_w, WF2, 1500, 128, KF2, NF2A);
    {
        dim3 g((128 + 63) / 64, (MFC + 255) / 256);
        gemm_bf16_wmma<<<g, 256, 0, stream>>>(G2B, WF2, out, fc2_b,
                                              MFC, NG, 128, KF2, KF2, KF2, 128, 0);
    }
}
